// MultiHeadCrossAttention_2241972929081
// MI455X (gfx1250) — compile-verified, matrix operand order checked by reading, not run on hardware
//
#include <hip/hip_runtime.h>

#define EMB  512
#define HEADS 8
#define HDIM 64
#define BSZ  2
#define SEQ  4096
#define BT   (BSZ*SEQ)

typedef _Float16 half_t;
typedef __attribute__((ext_vector_type(16))) _Float16 v16h;
typedef __attribute__((ext_vector_type(8)))  _Float16 v8h;
typedef __attribute__((ext_vector_type(8)))  float    v8f;
typedef __attribute__((ext_vector_type(4)))  float    v4f;
typedef __attribute__((ext_vector_type(4)))  int      v4i;

__device__ __forceinline__ v16h cat8(v8h lo, v8h hi) {
  return __builtin_shufflevector(lo, hi, 0,1,2,3,4,5,6,7,8,9,10,11,12,13,14,15);
}
__device__ __forceinline__ v8f wmma16(v16h a, v16h b, v8f c) {
  // D(16x16,f32) = A(16x32,f16) x B(32x16,f16) + C
  return __builtin_amdgcn_wmma_f32_16x16x32_f16(false, a, false, b, (short)0, c, false, false);
}

// ---------------------------------------------------------------------------
// GEMM: Out[M,512] = X[M,512] @ W[512,512]^T + bias   (torch Linear semantics)
// Computed transposed per tile: D = Wtile(A) x Xtile^T(B), so each lane owns
// one output row (m) and 8 consecutive output features -> packed b128 stores.
// Tile: 128 (M) x 64 (N) per workgroup, 8 waves, K-step 32, reg double-buffer.
// ---------------------------------------------------------------------------
template<bool IN_HALF, bool PERMUTE_OUT>
__global__ __launch_bounds__(256)
void gemm512_kernel(const void* __restrict__ Xv, const float* __restrict__ W,
                    const float* __restrict__ bias, void* __restrict__ Outv) {
  __shared__ __align__(16) half_t sA[128*32];   // X tile  [m][k]
  __shared__ __align__(16) half_t sB[64*32];    // W tile  [n][k]
  const int tid  = threadIdx.x;
  const int lane = tid & 31, wave = tid >> 5;
  const int ln   = lane & 15, hi16 = lane >> 4;
  const int koff = hi16 * 8;   // fragment K sub-offset (halves)
  const int rbas = hi16 * 8;   // D-tile row group
  const int m0 = blockIdx.x * 128;
  const int n0 = blockIdx.y * 64;

  const int rA = tid >> 1, chA = (tid & 1) * 16;   // X staging: 16 elems/thread
  const int rB = tid >> 2, chB = (tid & 3) * 8;    // W staging:  8 elems/thread

  v4f xa0, xa1, xa2, xa3;   // f32 input path
  v8h xh0, xh1;             // f16 input path
  v4f wb0, wb1;

  auto loadStage = [&](int k0) {
    if constexpr (IN_HALF) {
      const half_t* src = (const half_t*)Xv + (size_t)(m0 + rA) * EMB + k0 + chA;
      xh0 = *(const v8h*)src;
      xh1 = *(const v8h*)(src + 8);
    } else {
      const float* src = (const float*)Xv + (size_t)(m0 + rA) * EMB + k0 + chA;
      xa0 = *(const v4f*)(src);
      xa1 = *(const v4f*)(src + 4);
      xa2 = *(const v4f*)(src + 8);
      xa3 = *(const v4f*)(src + 12);
    }
    const float* ws = W + (size_t)(n0 + rB) * EMB + k0 + chB;
    wb0 = *(const v4f*)(ws);
    wb1 = *(const v4f*)(ws + 4);
  };
  auto storeStage = [&]() {
    if constexpr (IN_HALF) {
      *(v8h*)&sA[rA*32 + chA]     = xh0;
      *(v8h*)&sA[rA*32 + chA + 8] = xh1;
    } else {
      v8h h0, h1;
      #pragma unroll
      for (int j = 0; j < 4; ++j) {
        h0[j] = (half_t)xa0[j]; h0[4+j] = (half_t)xa1[j];
        h1[j] = (half_t)xa2[j]; h1[4+j] = (half_t)xa3[j];
      }
      *(v8h*)&sA[rA*32 + chA]     = h0;
      *(v8h*)&sA[rA*32 + chA + 8] = h1;
    }
    v8h hB;
    #pragma unroll
    for (int j = 0; j < 4; ++j) { hB[j] = (half_t)wb0[j]; hB[4+j] = (half_t)wb1[j]; }
    *(v8h*)&sB[rB*32 + chB] = hB;
  };

  v8f acc[4] = {};
  loadStage(0);

  for (int k0 = 0; k0 < EMB; k0 += 32) {
    __syncthreads();
    storeStage();
    __syncthreads();
    if (k0 + 32 < EMB) loadStage(k0 + 32);

    // B-fragment from X tile: lane = output row m' (col of D)
    v16h bX = cat8(*(const v8h*)&sA[(wave*16 + ln)*32 + koff],
                   *(const v8h*)&sA[(wave*16 + ln)*32 + koff + 16]);
    // A-fragments from W tile: lane = output feature row n
    #pragma unroll
    for (int nt = 0; nt < 4; ++nt) {
      v16h aW = cat8(*(const v8h*)&sB[(nt*16 + ln)*32 + koff],
                     *(const v8h*)&sB[(nt*16 + ln)*32 + koff + 16]);
      acc[nt] = wmma16(aW, bX, acc[nt]);   // D^T: rows = features, cols = m
    }
  }

  // ---- epilogue: lane owns row gr, 8 consecutive features per tile ----
  const int gr = m0 + wave*16 + ln;          // global row = b*T + t
  #pragma unroll
  for (int nt = 0; nt < 4; ++nt) {
    const int ebase = n0 + nt*16 + rbas;     // 8 consecutive output features
    v4f bv0 = *(const v4f*)(bias + ebase);
    v4f bv1 = *(const v4f*)(bias + ebase + 4);
    if constexpr (PERMUTE_OUT) {
      const int bidx = gr >> 12;             // / SEQ
      const int t    = gr & (SEQ - 1);
      const int h = ebase >> 6, d = ebase & 63;
      v8h hv;
      #pragma unroll
      for (int i = 0; i < 4; ++i) {
        hv[i]   = (half_t)(acc[nt][i]   + bv0[i]);
        hv[4+i] = (half_t)(acc[nt][4+i] + bv1[i]);
      }
      *(v8h*)&((half_t*)Outv)[(((size_t)bidx*HEADS + h)*SEQ + t)*HDIM + d] = hv;
    } else {
      float* orow = (float*)Outv + (size_t)gr*EMB + ebase;
      v4f o0, o1;
      #pragma unroll
      for (int i = 0; i < 4; ++i) { o0[i] = acc[nt][i] + bv0[i]; o1[i] = acc[nt][4+i] + bv1[i]; }
      *(v4f*)(orow)     = o0;
      *(v4f*)(orow + 4) = o1;
    }
  }
}

// ---------------------------------------------------------------------------
// Flash attention per (b, h, 128 q rows), 8 waves x 16 queries each.
// Transposed formulation: S^T = K Q^T  -> one query per lane (lane-local
// softmax, P^T is directly the B-fragment of O^T = V^T P^T, no LDS shuffle).
// ---------------------------------------------------------------------------
__global__ __launch_bounds__(256)
void attn_kernel(const half_t* __restrict__ Qh, const half_t* __restrict__ Kh,
                 const half_t* __restrict__ Vh, const int* __restrict__ mask,
                 half_t* __restrict__ Y) {
  __shared__ __align__(16) half_t sK[32*64];     // keys: [key][c]
  __shared__ __align__(16) half_t sVT[64*32];    // values transposed: [d][key]
  __shared__ __align__(16) float  sBias[128*32]; // mask -> additive bias (log2 dom)

  const int tid  = threadIdx.x;
  const int lane = tid & 31, wave = tid >> 5;
  const int ln   = lane & 15, hi16 = lane >> 4;
  const int koff = hi16 * 8;
  const int rbas = hi16 * 8;
  const int q0 = blockIdx.x * 128;
  const int h  = blockIdx.y;
  const int b  = blockIdx.z;

  const size_t headBase = ((size_t)b*HEADS + h) * (size_t)SEQ * HDIM;

  // Q as B-operand fragments (col = query = lane), kept in registers
  const half_t* qrow = Qh + headBase + (size_t)(q0 + wave*16 + ln) * HDIM;
  v16h bQ0 = cat8(*(const v8h*)(qrow + koff),      *(const v8h*)(qrow + koff + 16));
  v16h bQ1 = cat8(*(const v8h*)(qrow + 32 + koff), *(const v8h*)(qrow + 32 + koff + 16));

  v8f acc[4] = {};          // O^T: 4 d-tiles, this lane's query column
  float mrun = -3.0e38f, lrun = 0.0f;
  const float scl2 = 0.06376243208f;  // (1/sqrt(512)) * log2(e)

  // staging (register double-buffered)
  const int rKV = tid >> 3, cKV = (tid & 7) * 8;
  const int rM  = tid >> 1, cM  = (tid & 1) * 16;
  v8h kvr, vvr; v4i mr[4];
  auto loadStage = [&](int j0) {
    const half_t* kp = Kh + headBase + (size_t)(j0 + rKV)*HDIM + cKV;
    kvr = *(const v8h*)kp;
    vvr = *(const v8h*)(Vh + headBase + (size_t)(j0 + rKV)*HDIM + cKV);
    const int* mp = mask + ((size_t)b*SEQ + q0 + rM)*SEQ + j0 + cM;
    #pragma unroll
    for (int j = 0; j < 4; ++j) mr[j] = *(const v4i*)(mp + 4*j);
  };
  auto storeStage = [&]() {
    *(v8h*)&sK[rKV*64 + cKV] = kvr;
    #pragma unroll
    for (int j = 0; j < 8; ++j) sVT[(cKV + j)*32 + rKV] = vvr[j];
    #pragma unroll
    for (int j = 0; j < 4; ++j) {
      v4f bv4;
      #pragma unroll
      for (int u = 0; u < 4; ++u) bv4[u] = (mr[j][u] == 0) ? -1.44269504e10f : 0.0f;
      *(v4f*)&sBias[rM*32 + cM + 4*j] = bv4;
    }
  };

  loadStage(0);

  for (int j0 = 0; j0 < SEQ; j0 += 32) {
    __syncthreads();
    storeStage();
    __syncthreads();
    if (j0 + 32 < SEQ) loadStage(j0 + 32);

    // ---- S^T = K Q^T : st0 = keys 0..15, st1 = keys 16..31 (cols = queries)
    v8f st0 = {}, st1 = {};
    {
      v16h aK00 = cat8(*(const v8h*)&sK[ln*64 + koff],      *(const v8h*)&sK[ln*64 + koff + 16]);
      v16h aK10 = cat8(*(const v8h*)&sK[(16+ln)*64 + koff], *(const v8h*)&sK[(16+ln)*64 + koff + 16]);
      st0 = wmma16(aK00, bQ0, st0);
      st1 = wmma16(aK10, bQ0, st1);
      v16h aK01 = cat8(*(const v8h*)&sK[ln*64 + 32 + koff],      *(const v8h*)&sK[ln*64 + 32 + koff + 16]);
      v16h aK11 = cat8(*(const v8h*)&sK[(16+ln)*64 + 32 + koff], *(const v8h*)&sK[(16+ln)*64 + 32 + koff + 16]);
      st0 = wmma16(aK01, bQ1, st0);
      st1 = wmma16(aK11, bQ1, st1);
    }

    // ---- lane-local masked softmax (log2 domain), keys contiguous per lane
    const float* bp = &sBias[(wave*16 + ln)*32];
    v4f b00 = *(const v4f*)(bp + rbas);
    v4f b01 = *(const v4f*)(bp + rbas + 4);
    v4f b10 = *(const v4f*)(bp + 16 + rbas);
    v4f b11 = *(const v4f*)(bp + 16 + rbas + 4);
    float v0[8], v1[8];
    #pragma unroll
    for (int i = 0; i < 4; ++i) {
      v0[i]   = st0[i]  *scl2 + b00[i];
      v0[4+i] = st0[4+i]*scl2 + b01[i];
      v1[i]   = st1[i]  *scl2 + b10[i];
      v1[4+i] = st1[4+i]*scl2 + b11[i];
    }
    float vmax = v0[0];
    #pragma unroll
    for (int i = 1; i < 8; ++i) vmax = fmaxf(vmax, v0[i]);
    #pragma unroll
    for (int i = 0; i < 8; ++i) vmax = fmaxf(vmax, v1[i]);
    vmax = fmaxf(vmax, __shfl_xor(vmax, 16, 32));      // combine lane pair
    const float mnew = fmaxf(mrun, vmax);
    const float corr = exp2f(mrun - mnew);
    mrun = mnew;

    v16h bP;                       // P^T in C-layout == B-fragment layout
    float rs = 0.0f;
    #pragma unroll
    for (int i = 0; i < 8; ++i) { float p = exp2f(v0[i] - mnew); rs += p; bP[i]   = (half_t)p; }
    #pragma unroll
    for (int i = 0; i < 8; ++i) { float p = exp2f(v1[i] - mnew); rs += p; bP[8+i] = (half_t)p; }
    rs += __shfl_xor(rs, 16, 32);
    lrun = lrun*corr + rs;

    #pragma unroll
    for (int t = 0; t < 4; ++t) {
      #pragma unroll
      for (int i = 0; i < 8; ++i) acc[t][i] *= corr;
    }
    // ---- O^T += V^T P^T : 4 d-tiles, K = 32 keys in one WMMA each
    #pragma unroll
    for (int t = 0; t < 4; ++t) {
      const int d = t*16 + ln;
      v16h aV = cat8(*(const v8h*)&sVT[d*32 + koff], *(const v8h*)&sVT[d*32 + koff + 16]);
      acc[t] = wmma16(aV, bP, acc[t]);
    }
  }

  // ---- normalize + packed b128 stores (8 consecutive dims per lane/tile)
  const float inv = 1.0f / lrun;
  const int q = q0 + wave*16 + ln;
  half_t* yrow = Y + ((size_t)b*SEQ + q)*EMB + h*HDIM;
  #pragma unroll
  for (int t = 0; t < 4; ++t) {
    v8h hv;
    #pragma unroll
    for (int i = 0; i < 8; ++i) hv[i] = (half_t)(acc[t][i] * inv);
    *(v8h*)(yrow + t*16 + rbas) = hv;
  }
}

// ---------------------------------------------------------------------------
extern "C" void kernel_launch(void* const* d_in, const int* in_sizes, int n_in,
                              void* d_out, int out_size, void* d_ws, size_t ws_size,
                              hipStream_t stream) {
  (void)in_sizes; (void)n_in; (void)out_size; (void)ws_size;
  const float* x1  = (const float*)d_in[0];
  const float* x2  = (const float*)d_in[1];
  const int*  mask = (const int*)  d_in[2];
  const float* Wq  = (const float*)d_in[3];
  const float* bq  = (const float*)d_in[4];
  const float* Wk  = (const float*)d_in[5];
  const float* bk  = (const float*)d_in[6];
  const float* Wv  = (const float*)d_in[7];
  const float* bv  = (const float*)d_in[8];
  const float* Wu  = (const float*)d_in[9];
  const float* bu  = (const float*)d_in[10];

  // workspace: Q,K,V head-major f16 + Y row-major f16 (~33.6 MB)
  half_t* Qh = (half_t*)d_ws;
  const size_t seg = (size_t)BSZ*HEADS*SEQ*HDIM;
  half_t* Kh = Qh + seg;
  half_t* Vh = Kh + seg;
  half_t* Yh = Vh + seg;

  dim3 gG(BT/128, EMB/64);
  gemm512_kernel<false, true><<<gG, 256, 0, stream>>>(x1, Wq, bq, Qh);
  gemm512_kernel<false, true><<<gG, 256, 0, stream>>>(x2, Wk, bk, Kh);
  gemm512_kernel<false, true><<<gG, 256, 0, stream>>>(x2, Wv, bv, Vh);

  attn_kernel<<<dim3(SEQ/128, HEADS, BSZ), 256, 0, stream>>>(Qh, Kh, Vh, mask, Yh);

  gemm512_kernel<true, false><<<gG, 256, 0, stream>>>(Yh, Wu, bu, d_out);
}
